// ETFGraphModel_34479997452666
// MI455X (gfx1250) — matrix operand order repaired and checked by reading, not compiled, not run on hardware
//
#include <hip/hip_runtime.h>
#include <hip/hip_bf16.h>

// ---------------------------------------------------------------------------
// ETFGraphModel for MI455X (gfx1250, wave32, WMMA).
//   GAT GEMM (4096x512x128)  -> v_wmma_f32_16x16x32_f16, 16x64 tile/wave,
//                               b128 vector operand loads (W pre-transposed)
//   edge softmax/aggregate   -> atomics (ordered-uint max for segment_max)
//   LSTM scan (serial, 4096) -> single WGP, LDS-resident state
//   corr = xn @ xn^T (4096x4096x128) -> 64x64 macro-tile/wave: 4x4 WMMA
//                               accumulators (128 VGPRs), 16 wmma / K-chunk
//   MLP head                 -> wave-per-node, shfl reduction
// ---------------------------------------------------------------------------

typedef __attribute__((ext_vector_type(16))) _Float16 v16h;
typedef __attribute__((ext_vector_type(8)))  _Float16 v8h;
typedef __attribute__((ext_vector_type(8)))  float    v8f;

#define N_NODES 4096
#define D_INF   128
#define HDIM    128
#define HEADS   4
#define NHTOT   (HEADS * HDIM)   // 512
#define NEG_SLOPE 0.2f

// ---------------- helpers -------------------------------------------------

__device__ __forceinline__ unsigned f2key(float f) {
  unsigned u = __float_as_uint(f);
  return (u & 0x80000000u) ? ~u : (u | 0x80000000u);
}
__device__ __forceinline__ float key2f(unsigned k) {
  return __uint_as_float((k & 0x80000000u) ? (k ^ 0x80000000u) : ~k);
}
__device__ __forceinline__ float sigmoidf_dev(float x) {
  return 1.0f / (1.0f + __expf(-x));
}

// Build a v16h WMMA operand from two contiguous 8-half (16-byte) chunks.
// A operand: chunks at p and p+16 (element k-split 0..7 | 16..23).
// B operand (K-contiguous storage): chunks at p and p+16 (same k-split as A).
__device__ __forceinline__ v16h ld2x8(const _Float16* __restrict__ p, int gap) {
  v8h lo = *(const v8h*)(p);
  v8h hi = *(const v8h*)(p + gap);
  v16h r;
#pragma unroll
  for (int i = 0; i < 8; ++i) { r[i] = lo[i]; r[8 + i] = hi[i]; }
  return r;
}

// f32 -> f16 convert
__global__ void f32_to_f16_kernel(const float* __restrict__ in,
                                  _Float16* __restrict__ out, int n) {
  int i = blockIdx.x * blockDim.x + threadIdx.x;
  if (i < n) out[i] = (_Float16)in[i];
}

// f32 (K x N) -> f16 transposed (N x K), K-contiguous rows for B operands
__global__ void transpose_to_f16_kernel(const float* __restrict__ in,
                                        _Float16* __restrict__ out,
                                        int K, int N) {
  int idx = blockIdx.x * blockDim.x + threadIdx.x;
  if (idx >= K * N) return;
  int k = idx / N, n = idx % N;
  out[n * K + k] = (_Float16)in[idx];
}

// ---------------- GAT GEMM: H = x @ W_gat  (M=4096, K=128, N=512) ---------
// 16x64 tile per wave: 1 A tile reused over 4 B tiles, 4 f32 accumulators.
__global__ void __launch_bounds__(256)
gat_gemm_kernel(const _Float16* __restrict__ A16,   // x   (4096 x 128)
                const _Float16* __restrict__ BT16,  // W^T (512 x 128), K-contig
                float* __restrict__ H) {            // out (4096 x 512)
  int wave = blockIdx.x * (blockDim.x >> 5) + (threadIdx.x >> 5);
  int lane = threadIdx.x & 31;
  int tm = wave >> 3;                 // 0..255  (4096/16)
  int tn = wave & 7;                  // 0..7    (512/64)
  int mBase = tm * 16, nBase = tn * 64;
  int lrow = lane & 15, half = lane >> 4;

  v8f acc[4] = {};
#pragma unroll
  for (int kc = 0; kc < 128; kc += 32) {
    v16h a = ld2x8(A16 + (mBase + lrow) * 128 + kc + (half << 3), 16);
#pragma unroll
    for (int j = 0; j < 4; ++j) {
      v16h b = ld2x8(BT16 + (nBase + j * 16 + lrow) * 128 + kc + (half << 3), 16);
      acc[j] = __builtin_amdgcn_wmma_f32_16x16x32_f16(false, a, false, b,
                                                      (short)0, acc[j],
                                                      false, false);
    }
  }
#pragma unroll
  for (int j = 0; j < 4; ++j)
#pragma unroll
    for (int r = 0; r < 8; ++r) {
      int m = mBase + r + (half << 3);
      int n = nBase + j * 16 + lrow;
      H[m * 512 + n] = acc[j][r];
    }
}

// ---------------- attention scores per (node, head) -----------------------
__global__ void att_kernel(const float* __restrict__ H,
                           const float* __restrict__ att_src,
                           const float* __restrict__ att_dst,
                           float* __restrict__ a_src,
                           float* __restrict__ a_dst) {
  int idx = blockIdx.x * blockDim.x + threadIdx.x;   // N*HEADS
  if (idx >= N_NODES * HEADS) return;
  int n = idx >> 2, hd = idx & 3;
  const float* hrow = H + n * NHTOT + hd * HDIM;
  float as = 0.f, ad = 0.f;
#pragma unroll 4
  for (int k = 0; k < HDIM; ++k) {
    as += hrow[k] * att_src[hd * HDIM + k];
    ad += hrow[k] * att_dst[hd * HDIM + k];
  }
  a_src[idx] = as;
  a_dst[idx] = ad;
}

// ---------------- init accumulators ---------------------------------------
__global__ void init_kernel(float* __restrict__ agg, float* __restrict__ ssum,
                            unsigned* __restrict__ mkey, int aggN, int smN) {
  int i = blockIdx.x * blockDim.x + threadIdx.x;
  if (i < aggN) agg[i] = 0.f;
  if (i < smN) { ssum[i] = 0.f; mkey[i] = 0u; }   // key 0 == most-negative
}

__device__ __forceinline__ void edge_endpoints(const int* __restrict__ ei,
                                               int E, int e, int& s, int& d) {
  if (e < E) { s = ei[e]; d = ei[E + e]; }
  else       { s = d = e - E; }                    // appended self-loops
}

// ---------------- edge pass 1: leaky-relu + segment max -------------------
__global__ void edge_max_kernel(const int* __restrict__ ei, int E, int ET,
                                const float* __restrict__ a_src,
                                const float* __restrict__ a_dst,
                                float* __restrict__ eval,
                                unsigned* __restrict__ mkey) {
  int idx = blockIdx.x * blockDim.x + threadIdx.x;   // ET*HEADS
  if (idx >= ET * HEADS) return;
  int e = idx >> 2, hd = idx & 3;
  int s, d; edge_endpoints(ei, E, e, s, d);
  float v = a_src[s * HEADS + hd] + a_dst[d * HEADS + hd];
  v = (v > 0.f) ? v : NEG_SLOPE * v;
  eval[idx] = v;
  atomicMax(&mkey[d * HEADS + hd], f2key(v));
}

// ---------------- edge pass 2: exp + segment sum --------------------------
__global__ void edge_sum_kernel(const int* __restrict__ ei, int E, int ET,
                                const float* __restrict__ eval,
                                const unsigned* __restrict__ mkey,
                                float* __restrict__ ee,
                                float* __restrict__ ssum) {
  int idx = blockIdx.x * blockDim.x + threadIdx.x;
  if (idx >= ET * HEADS) return;
  int e = idx >> 2, hd = idx & 3;
  int s, d; edge_endpoints(ei, E, e, s, d);
  (void)s;
  float m = key2f(mkey[d * HEADS + hd]);
  float x = __expf(eval[idx] - m);
  ee[idx] = x;
  atomicAdd(&ssum[d * HEADS + hd], x);
}

// ---------------- edge pass 3: weighted aggregation -----------------------
// one block (128 threads) per edge; each thread owns one channel, loops heads
__global__ void __launch_bounds__(128)
edge_agg_kernel(const int* __restrict__ ei, int E, int ET,
                const float* __restrict__ H,
                const float* __restrict__ ee,
                const float* __restrict__ ssum,
                float* __restrict__ agg) {
  int e = blockIdx.x;
  if (e >= ET) return;
  int c = threadIdx.x;
  int s, d; edge_endpoints(ei, E, e, s, d);
#pragma unroll
  for (int hd = 0; hd < HEADS; ++hd) {
    float alpha = ee[e * HEADS + hd] / (ssum[d * HEADS + hd] + 1e-16f);
    atomicAdd(&agg[d * NHTOT + hd * HDIM + c],
              alpha * H[s * NHTOT + hd * HDIM + c]);
  }
}

// ---------------- GAT finish: mean over heads + bias + relu ---------------
__global__ void gat_finish_kernel(const float* __restrict__ agg,
                                  const float* __restrict__ b_gat,
                                  float* __restrict__ g) {
  int idx = blockIdx.x * blockDim.x + threadIdx.x;   // N*HDIM
  if (idx >= N_NODES * HDIM) return;
  int n = idx >> 7, c = idx & 127;
  float sum = 0.f;
#pragma unroll
  for (int hd = 0; hd < HEADS; ++hd) sum += agg[n * NHTOT + hd * HDIM + c];
  float v = sum * 0.25f + b_gat[c];
  g[idx] = (v > 0.f) ? v : 0.f;
}

// ---------------- LSTM scan (serial): one 512-thread workgroup ------------
__global__ void __launch_bounds__(512)
lstm_scan_kernel(const float* __restrict__ g,
                 const float* __restrict__ W_ih, const float* __restrict__ W_hh,
                 const float* __restrict__ b_ih, const float* __restrict__ b_hh,
                 float* __restrict__ emb) {
  __shared__ float sh[HDIM], sc[HDIM], sx[HDIM], sg[4 * HDIM];
  int t = threadIdx.x;                 // 0..511, one gate row each
  if (t < HDIM) { sh[t] = 0.f; sc[t] = 0.f; }
  __syncthreads();
  const float* wi = W_ih + t * HDIM;
  const float* wh = W_hh + t * HDIM;
  float bias = b_ih[t] + b_hh[t];
  for (int step = 0; step < N_NODES; ++step) {
    if (t < HDIM) sx[t] = g[step * HDIM + t];
    __syncthreads();
    float acc = bias;
#pragma unroll 4
    for (int k = 0; k < HDIM; k += 4) {
      acc += wi[k] * sx[k] + wi[k + 1] * sx[k + 1] +
             wi[k + 2] * sx[k + 2] + wi[k + 3] * sx[k + 3];
      acc += wh[k] * sh[k] + wh[k + 1] * sh[k + 1] +
             wh[k + 2] * sh[k + 2] + wh[k + 3] * sh[k + 3];
    }
    sg[t] = acc;
    __syncthreads();
    if (t < HDIM) {
      float ig = sigmoidf_dev(sg[t]);
      float fg = sigmoidf_dev(sg[HDIM + t]);
      float gg = tanhf(sg[2 * HDIM + t]);
      float og = sigmoidf_dev(sg[3 * HDIM + t]);
      float c = fg * sc[t] + ig * gg;
      float h = og * tanhf(c);
      sc[t] = c; sh[t] = h;
      emb[step * HDIM + t] = h;
    }
    __syncthreads();
  }
}

// ---------------- row normalize -> f16 ------------------------------------
__global__ void __launch_bounds__(128)
normalize_kernel(const float* __restrict__ emb, _Float16* __restrict__ xn16) {
  __shared__ float red[HDIM];
  int n = blockIdx.x, t = threadIdx.x;
  float v = emb[n * HDIM + t];
  red[t] = v * v;
  __syncthreads();
  for (int s2 = 64; s2 > 0; s2 >>= 1) {
    if (t < s2) red[t] += red[t + s2];
    __syncthreads();
  }
  float nrm = fmaxf(sqrtf(red[0]), 1e-12f);
  xn16[n * HDIM + t] = (_Float16)(v / nrm);
}

// ---------------- corr = xn @ xn^T (4096x4096x128) ------------------------
// 64x64 macro-tile per wave: 4x4 WMMA tiles, 16 f32 accumulators (128 VGPRs).
// A and B operands are both rows of xn (B = xn^T loaded column-major), so all
// operand loads are pairs of contiguous global_load_b128.
__global__ void __launch_bounds__(256)
corr_gemm_kernel(const _Float16* __restrict__ XN, float* __restrict__ C) {
  int wave = blockIdx.x * (blockDim.x >> 5) + (threadIdx.x >> 5);
  int lane = threadIdx.x & 31;
  int tm = wave >> 6;                  // 0..63  (4096/64)
  int tn = wave & 63;
  int mBase = tm * 64, nBase = tn * 64;
  int lrow = lane & 15, half = lane >> 4;

  v8f acc[4][4] = {};
#pragma unroll
  for (int kc = 0; kc < 128; kc += 32) {
    v16h a[4], b[4];
#pragma unroll
    for (int t = 0; t < 4; ++t) {
      a[t] = ld2x8(XN + (mBase + t * 16 + lrow) * 128 + kc + (half << 3), 16);
      b[t] = ld2x8(XN + (nBase + t * 16 + lrow) * 128 + kc + (half << 3), 16);
    }
#pragma unroll
    for (int ti = 0; ti < 4; ++ti)
#pragma unroll
      for (int tj = 0; tj < 4; ++tj)
        acc[ti][tj] = __builtin_amdgcn_wmma_f32_16x16x32_f16(
            false, a[ti], false, b[tj], (short)0, acc[ti][tj], false, false);
  }
#pragma unroll
  for (int ti = 0; ti < 4; ++ti)
#pragma unroll
    for (int tj = 0; tj < 4; ++tj)
#pragma unroll
      for (int r = 0; r < 8; ++r) {
        int m = mBase + ti * 16 + r + (half << 3);
        int n = nBase + tj * 16 + lrow;
        C[m * 4096 + n] = acc[ti][tj][r];
      }
}

// ---------------- MLP head: wave per node ---------------------------------
__global__ void __launch_bounds__(256)
mlp_kernel(const float* __restrict__ emb,
           const float* __restrict__ W1, const float* __restrict__ b1,
           const float* __restrict__ W2, const float* __restrict__ b2,
           float* __restrict__ mu, float* __restrict__ sigma) {
  int node = blockIdx.x * (blockDim.x >> 5) + (threadIdx.x >> 5);
  int lane = threadIdx.x & 31;
  if (node >= N_NODES) return;
  const float* e = emb + node * HDIM;
  float hidden = b1[lane];
#pragma unroll 4
  for (int k = 0; k < HDIM; ++k) hidden += e[k] * W1[k * 32 + lane];
  hidden = (hidden > 0.f) ? hidden : 0.f;
  float o0 = hidden * W2[lane * 2 + 0];
  float o1 = hidden * W2[lane * 2 + 1];
  for (int off = 16; off > 0; off >>= 1) {
    o0 += __shfl_xor(o0, off);
    o1 += __shfl_xor(o1, off);
  }
  if (lane == 0) {
    mu[node] = o0 + b2[0];
    sigma[node] = o1 + b2[1];
  }
}

// ---------------------------------------------------------------------------
extern "C" void kernel_launch(void* const* d_in, const int* in_sizes, int n_in,
                              void* d_out, int out_size, void* d_ws, size_t ws_size,
                              hipStream_t stream) {
  const float* x       = (const float*)d_in[0];
  const int*   ei      = (const int*)d_in[1];
  const float* W_gat   = (const float*)d_in[2];
  const float* att_src = (const float*)d_in[3];
  const float* att_dst = (const float*)d_in[4];
  const float* b_gat   = (const float*)d_in[5];
  const float* W_ih    = (const float*)d_in[6];
  const float* W_hh    = (const float*)d_in[7];
  const float* b_ih    = (const float*)d_in[8];
  const float* b_hh    = (const float*)d_in[9];
  const float* W1      = (const float*)d_in[10];
  const float* b1      = (const float*)d_in[11];
  const float* W2      = (const float*)d_in[12];
  const float* b2      = (const float*)d_in[13];

  const int E  = in_sizes[1] / 2;
  const int ET = E + N_NODES;

  // outputs: emb (N*HD) | corr (N*N) | mu (N) | sigma (N)
  float* out_emb   = (float*)d_out;
  float* out_corr  = out_emb + N_NODES * HDIM;
  float* out_mu    = out_corr + (size_t)N_NODES * N_NODES;
  float* out_sigma = out_mu + N_NODES;

  // workspace carve-up (~26 MB)
  char* w = (char*)d_ws;
  auto carve = [&](size_t bytes) {
    char* p = w;
    w += (bytes + 255) & ~(size_t)255;
    return p;
  };
  _Float16* xf16  = (_Float16*)carve((size_t)N_NODES * D_INF * 2);
  _Float16* wT16  = (_Float16*)carve((size_t)D_INF * NHTOT * 2);  // 512 x 128
  float*    H     = (float*)carve((size_t)N_NODES * NHTOT * 4);
  float*    a_src = (float*)carve((size_t)N_NODES * HEADS * 4);
  float*    a_dst = (float*)carve((size_t)N_NODES * HEADS * 4);
  unsigned* mkey  = (unsigned*)carve((size_t)N_NODES * HEADS * 4);
  float*    ssum  = (float*)carve((size_t)N_NODES * HEADS * 4);
  float*    eval  = (float*)carve((size_t)ET * HEADS * 4);
  float*    ee    = (float*)carve((size_t)ET * HEADS * 4);
  float*    agg   = (float*)carve((size_t)N_NODES * NHTOT * 4);
  float*    g     = (float*)carve((size_t)N_NODES * HDIM * 4);
  _Float16* xn16  = (_Float16*)carve((size_t)N_NODES * HDIM * 2);

  // 1) f32 -> f16 conversions (x plain; W_gat transposed to K-contiguous)
  {
    int n1 = N_NODES * D_INF;
    f32_to_f16_kernel<<<(n1 + 255) / 256, 256, 0, stream>>>(x, xf16, n1);
    int n2 = D_INF * NHTOT;
    transpose_to_f16_kernel<<<(n2 + 255) / 256, 256, 0, stream>>>(
        W_gat, wT16, D_INF, NHTOT);
  }

  // 2) H = x @ W_gat via WMMA  (2048 waves of 16x64 tiles, 8 waves/block)
  gat_gemm_kernel<<<256, 256, 0, stream>>>(xf16, wT16, H);

  // 3) attention scores
  att_kernel<<<(N_NODES * HEADS + 255) / 256, 256, 0, stream>>>(
      H, att_src, att_dst, a_src, a_dst);

  // 4) init segment accumulators
  {
    int aggN = N_NODES * NHTOT, smN = N_NODES * HEADS;
    init_kernel<<<(aggN + 255) / 256, 256, 0, stream>>>(agg, ssum, mkey, aggN, smN);
  }

  // 5-7) edge softmax + aggregation
  {
    int nEdgeWork = ET * HEADS;
    edge_max_kernel<<<(nEdgeWork + 255) / 256, 256, 0, stream>>>(
        ei, E, ET, a_src, a_dst, eval, mkey);
    edge_sum_kernel<<<(nEdgeWork + 255) / 256, 256, 0, stream>>>(
        ei, E, ET, eval, mkey, ee, ssum);
    edge_agg_kernel<<<ET, 128, 0, stream>>>(ei, E, ET, H, ee, ssum, agg);
  }

  // 8) mean over heads + bias + relu
  gat_finish_kernel<<<(N_NODES * HDIM + 255) / 256, 256, 0, stream>>>(agg, b_gat, g);

  // 9) serial LSTM scan -> emb (written straight into d_out)
  lstm_scan_kernel<<<1, 512, 0, stream>>>(g, W_ih, W_hh, b_ih, b_hh, out_emb);

  // 10) row-normalize -> f16 operand for corr GEMM
  normalize_kernel<<<N_NODES, 128, 0, stream>>>(out_emb, xn16);

  // 11) corr = xn @ xn^T via WMMA  (4096 waves of 64x64 macro-tiles)
  corr_gemm_kernel<<<512, 256, 0, stream>>>(xn16, out_corr);

  // 12) MLP head -> mu, sigma
  mlp_kernel<<<(N_NODES + 7) / 8, 256, 0, stream>>>(
      out_emb, W1, b1, W2, b2, out_mu, out_sigma);

  (void)n_in; (void)out_size; (void)ws_size;
}
